// MultiHeadAttention_21234318312137
// MI455X (gfx1250) — matrix operand order repaired and checked by reading, not compiled, not run on hardware
//
#include <hip/hip_runtime.h>
#include <hip/hip_bf16.h>
#include <stdint.h>

// ---------------- problem constants ----------------
constexpr int BATCH  = 4;
constexpr int TSEQ   = 2048;
constexpr int DMODEL = 1024;
constexpr int NHEAD  = 16;
constexpr int DHEAD  = 64;
constexpr int MROWS  = BATCH * TSEQ;  // 8192

typedef __bf16 v16bf __attribute__((ext_vector_type(16)));
typedef __bf16 v8bf  __attribute__((ext_vector_type(8)));
typedef float  v8f   __attribute__((ext_vector_type(8)));
typedef float  v4f   __attribute__((ext_vector_type(4)));

// ---------------- WMMA wrapper (CDNA5 bf16, f32 accumulate) ----------------
__device__ __forceinline__ v8f wmma_bf16(v16bf a, v16bf b, v8f c) {
  // (neg_a, A, neg_b, B, c_mod, C, reuse_a, reuse_b)
  return __builtin_amdgcn_wmma_f32_16x16x32_bf16(false, a, false, b, (short)0, c,
                                                 false, false);
}

__device__ __forceinline__ float fast_exp2(float x) {
#if __has_builtin(__builtin_amdgcn_exp2f)
  return __builtin_amdgcn_exp2f(x);   // v_exp_f32
#else
  return exp2f(x);
#endif
}

// ---------------- async global -> LDS (CDNA5) ----------------
#if defined(__gfx1250__) && __has_builtin(__builtin_amdgcn_global_load_async_to_lds_b128)
#define USE_ASYNC_LDS 1
#else
#define USE_ASYNC_LDS 0
#endif

#if USE_ASYNC_LDS
// Builtin expects pointers to a 16-byte int vector in addrspace(1)/(3).
typedef int b128_t __attribute__((vector_size(16)));
typedef __attribute__((address_space(1))) b128_t* gptr128_t;
typedef __attribute__((address_space(3))) b128_t* lptr128_t;
#endif

__device__ __forceinline__ void copy16_to_lds(const __bf16* g, __bf16* l) {
#if USE_ASYNC_LDS
  __builtin_amdgcn_global_load_async_to_lds_b128((gptr128_t)(g), (lptr128_t)(l),
                                                 0, 0);
#else
  *(v8bf*)l = *(const v8bf*)g;
#endif
}

// ======================================================================
// Kernel 1: W[k][n] f32  ->  Wt[n][k] bf16  (for all three weights)
// ======================================================================
__global__ __launch_bounds__(256) void mha_wt_transpose(
    const float* __restrict__ W0, const float* __restrict__ W1,
    const float* __restrict__ W2, __bf16* __restrict__ T0,
    __bf16* __restrict__ T1, __bf16* __restrict__ T2) {
  const int mat = blockIdx.z;
  const float* W = (mat == 0) ? W0 : (mat == 1) ? W1 : W2;
  __bf16* Tt     = (mat == 0) ? T0 : (mat == 1) ? T1 : T2;
  const int idx = blockIdx.x * 256 + threadIdx.x;  // < 1024*1024
  const int kk = idx >> 10;
  const int nn = idx & (DMODEL - 1);
  Tt[(size_t)nn * DMODEL + kk] = (__bf16)W[idx];
}

// ======================================================================
// Kernel 2: QKV projection GEMM.  X[8192x1024] f32 * Wt^T -> heads (bf16)
//   mat 0 -> Qh[b][h][t][64], mat 1 -> Kh[b][h][t][64],
//   mat 2 -> VhT[b][h][64][t]  (transposed for the P*V B-fragments)
// Wave computes a 32x64 tile (2x4 WMMA frags); WG = 8 waves = 128x128.
// ======================================================================
__device__ __forceinline__ v16bf load_a_f32(const float* rowp, int klo, int khi) {
  v4f a0 = *(const v4f*)(rowp + klo);
  v4f a1 = *(const v4f*)(rowp + klo + 4);
  v4f a2 = *(const v4f*)(rowp + khi);
  v4f a3 = *(const v4f*)(rowp + khi + 4);
  v16bf r;
#pragma unroll
  for (int i = 0; i < 4; ++i) {
    r[i]      = (__bf16)a0[i];
    r[4 + i]  = (__bf16)a1[i];
    r[8 + i]  = (__bf16)a2[i];
    r[12 + i] = (__bf16)a3[i];
  }
  return r;
}

__global__ __launch_bounds__(256) void mha_qkv_proj(
    const float* __restrict__ q, const float* __restrict__ k,
    const float* __restrict__ v, const __bf16* __restrict__ WtQ,
    const __bf16* __restrict__ WtK, const __bf16* __restrict__ WtV,
    __bf16* __restrict__ Qh, __bf16* __restrict__ Kh,
    __bf16* __restrict__ VhT) {
  const int tid  = threadIdx.x;
  const int lane = tid & 31;
  const int wave = tid >> 5;
  const int half = lane >> 4;
  const int hr   = lane & 15;
  const int mat  = blockIdx.z;

  const int m0 = blockIdx.x * 128 + (wave & 3) * 32;
  const int n0 = blockIdx.y * 128 + (wave >> 2) * 64;

  const float*  X  = (mat == 0) ? q : (mat == 1) ? k : v;
  const __bf16* Wt = (mat == 0) ? WtQ : (mat == 1) ? WtK : WtV;

  const v8f vzero = {0, 0, 0, 0, 0, 0, 0, 0};
  v8f acc[2][4];
#pragma unroll
  for (int i = 0; i < 2; ++i)
#pragma unroll
    for (int j = 0; j < 4; ++j) acc[i][j] = vzero;

  const float* rowA0 = X + (size_t)(m0 + hr) * DMODEL;
  const float* rowA1 = X + (size_t)(m0 + 16 + hr) * DMODEL;

  for (int kk = 0; kk < DMODEL; kk += 32) {
    // A fragments: lanes 0-15 K in [kk, kk+16); lanes 16-31 K in [kk+16, kk+32)
    v16bf A0 = load_a_f32(rowA0, kk + half * 16, kk + half * 16 + 8);
    v16bf A1 = load_a_f32(rowA1, kk + half * 16, kk + half * 16 + 8);
    // B fragments from transposed weights: contiguous 16 bf16 along K per lane
    v16bf Bf[4];
#pragma unroll
    for (int nf = 0; nf < 4; ++nf)
      Bf[nf] = *(const v16bf*)(Wt + (size_t)(n0 + nf * 16 + hr) * DMODEL + kk +
                               half * 16);
#pragma unroll
    for (int nf = 0; nf < 4; ++nf) {
      acc[0][nf] = wmma_bf16(A0, Bf[nf], acc[0][nf]);
      acc[1][nf] = wmma_bf16(A1, Bf[nf], acc[1][nf]);
    }
  }

  // epilogue: scatter into head-split bf16 buffers
#pragma unroll
  for (int mf = 0; mf < 2; ++mf) {
    const int rowbase = m0 + mf * 16 + half * 8;  // C layout: VGPR j -> row j / 8+j
#pragma unroll
    for (int nf = 0; nf < 4; ++nf) {
      const int col  = n0 + nf * 16 + hr;
      const int hcol = col >> 6;
      const int dcol = col & (DHEAD - 1);
#pragma unroll
      for (int j = 0; j < 8; ++j) {
        const int grow = rowbase + j;
        const int bb   = grow >> 11;           // / TSEQ
        const int tt   = grow & (TSEQ - 1);
        const __bf16 val = (__bf16)acc[mf][nf][j];
        if (mat == 0)
          Qh[(((size_t)bb * NHEAD + hcol) * TSEQ + tt) * DHEAD + dcol] = val;
        else if (mat == 1)
          Kh[(((size_t)bb * NHEAD + hcol) * TSEQ + tt) * DHEAD + dcol] = val;
        else
          VhT[(((size_t)bb * NHEAD + hcol) * DHEAD + dcol) * TSEQ + tt] = val;
      }
    }
  }
}

// ======================================================================
// Kernel 3: flash attention + residual.
// Grid (T/128, NHEAD, BATCH); 8 waves x 16 q-rows = 128 q-rows per WG.
// K/V 32-key tiles double-buffered in LDS via async global->LDS copies.
// ======================================================================
__global__ __launch_bounds__(256) void mha_attention(
    const __bf16* __restrict__ Qh, const __bf16* __restrict__ Kh,
    const __bf16* __restrict__ VhT, const float* __restrict__ resid,
    float* __restrict__ out) {
  const int tid  = threadIdx.x;
  const int lane = tid & 31;
  const int wave = tid >> 5;
  const int half = lane >> 4;
  const int hr   = lane & 15;

  const int b  = blockIdx.z;
  const int h  = blockIdx.y;
  const int q0 = blockIdx.x * 128 + wave * 16;
  const int bh = b * NHEAD + h;

  const __bf16* Qp = Qh + ((size_t)bh * TSEQ + q0) * DHEAD;
  const __bf16* Kp = Kh + (size_t)bh * TSEQ * DHEAD;
  const __bf16* Vp = VhT + (size_t)bh * DHEAD * TSEQ;

  __shared__ alignas(32) __bf16 Ks[2][32 * DHEAD];   // [key][d]
  __shared__ alignas(32) __bf16 Vs[2][DHEAD * 32];   // [d][key]
  __shared__ alignas(32) __bf16 Ps[8][16 * 32];      // per-wave P relayout

  // Q A-fragments (whole kernel lifetime): d-chunks [0,32) and [32,64)
  v16bf qa[2];
#pragma unroll
  for (int f = 0; f < 2; ++f) {
    const __bf16* qrow = Qp + (size_t)hr * DHEAD + f * 32;
    v8bf lo = *(const v8bf*)(qrow + half * 16);
    v8bf hi = *(const v8bf*)(qrow + half * 16 + 8);
    v16bf a;
#pragma unroll
    for (int i = 0; i < 8; ++i) { a[i] = lo[i]; a[8 + i] = hi[i]; }
    qa[f] = a;
  }

  float mrow[8], lrow[8];
#pragma unroll
  for (int j = 0; j < 8; ++j) { mrow[j] = -1e30f; lrow[j] = 0.0f; }
  const v8f vzero = {0, 0, 0, 0, 0, 0, 0, 0};
  v8f Oc[4] = {vzero, vzero, vzero, vzero};

  constexpr int NIT = TSEQ / 32;

  auto issue = [&](int it) {
    const int buf = it & 1;
    const int kt  = it * 32;
    // K tile: 32 keys x 64 d (bf16, row-major) = 4KB, 256 threads x 16B
    copy16_to_lds(Kp + (size_t)(kt + (tid >> 3)) * DHEAD + (tid & 7) * 8,
                  &Ks[buf][(tid >> 3) * DHEAD + (tid & 7) * 8]);
    // V tile (pre-transposed): 64 d x 32 keys = 4KB
    copy16_to_lds(Vp + (size_t)(tid >> 2) * TSEQ + kt + (tid & 3) * 8,
                  &Vs[buf][(tid >> 2) * 32 + (tid & 3) * 8]);
  };

  issue(0);
  issue(1);

  __bf16* Pw = &Ps[wave][0];
  const float cs = 0.0450842200f;  // log2(e) / sqrt(d_model=1024)

  for (int it = 0; it < NIT; ++it) {
    const int buf = it & 1;
#if USE_ASYNC_LDS
    if (it + 1 < NIT) asm volatile("s_wait_asynccnt 0x2" ::: "memory");
    else              asm volatile("s_wait_asynccnt 0x0" ::: "memory");
#endif
    __syncthreads();

    const __bf16* Kbuf = &Ks[buf][0];
    const __bf16* Vbuf = &Vs[buf][0];

    // ---- S = Q(16x64) * K^T(64x32) : two 16x16 f32 fragments ----
    v8f s0 = vzero, s1 = vzero;
    {
      v16bf b00 = *(const v16bf*)(Kbuf + (hr)*DHEAD + half * 16);
      v16bf b01 = *(const v16bf*)(Kbuf + (hr)*DHEAD + 32 + half * 16);
      s0 = wmma_bf16(qa[0], b00, s0);
      s0 = wmma_bf16(qa[1], b01, s0);
      v16bf b10 = *(const v16bf*)(Kbuf + (16 + hr) * DHEAD + half * 16);
      v16bf b11 = *(const v16bf*)(Kbuf + (16 + hr) * DHEAD + 32 + half * 16);
      s1 = wmma_bf16(qa[0], b10, s1);
      s1 = wmma_bf16(qa[1], b11, s1);
    }

    // ---- online softmax (base-2, scale folded) ----
#pragma unroll
    for (int j = 0; j < 8; ++j) {
      float a0 = s0[j] * cs, a1 = s1[j] * cs;
      float mx = fmaxf(a0, a1);
#pragma unroll
      for (int d = 1; d < 16; d <<= 1) mx = fmaxf(mx, __shfl_xor(mx, d));
      float mnew  = fmaxf(mrow[j], mx);
      float alpha = fast_exp2(mrow[j] - mnew);
      float p0    = fast_exp2(a0 - mnew);
      float p1    = fast_exp2(a1 - mnew);
      float rs    = p0 + p1;
#pragma unroll
      for (int d = 1; d < 16; d <<= 1) rs += __shfl_xor(rs, d);
      lrow[j] = lrow[j] * alpha + rs;
      mrow[j] = mnew;
#pragma unroll
      for (int nf = 0; nf < 4; ++nf) Oc[nf][j] *= alpha;
      // stash P in [row][key] order for A-fragment reload
      Pw[(half * 8 + j) * 32 + hr]      = (__bf16)p0;
      Pw[(half * 8 + j) * 32 + 16 + hr] = (__bf16)p1;
    }

    // ---- reload P as A-fragment (16x32) ----
    v16bf pa;
    {
      v8bf lo = *(const v8bf*)(Pw + hr * 32 + half * 16);
      v8bf hi = *(const v8bf*)(Pw + hr * 32 + half * 16 + 8);
#pragma unroll
      for (int i = 0; i < 8; ++i) { pa[i] = lo[i]; pa[8 + i] = hi[i]; }
    }

    // ---- O(16x64) += P(16x32) * V(32x64) ----
#pragma unroll
    for (int nf = 0; nf < 4; ++nf) {
      v16bf bv = *(const v16bf*)(Vbuf + (nf * 16 + hr) * 32 + half * 16);
      Oc[nf] = wmma_bf16(pa, bv, Oc[nf]);
    }

    __syncthreads();
    if (it + 2 < NIT) issue(it + 2);
  }

  // ---- epilogue: normalize, add residual, store f32 ----
  const float* resp = resid + (size_t)b * TSEQ * DMODEL + h * DHEAD;
  float*       outp = out + (size_t)b * TSEQ * DMODEL + h * DHEAD;
#pragma unroll
  for (int j = 0; j < 8; ++j) {
    const int t = q0 + half * 8 + j;
    const float invl = 1.0f / lrow[j];
    const size_t roff = (size_t)t * DMODEL;
#pragma unroll
    for (int nf = 0; nf < 4; ++nf) {
      const size_t idx = roff + nf * 16 + hr;
      outp[idx] = resp[idx] + Oc[nf][j] * invl;
    }
  }
}

// ======================================================================
extern "C" void kernel_launch(void* const* d_in, const int* in_sizes, int n_in,
                              void* d_out, int out_size, void* d_ws,
                              size_t ws_size, hipStream_t stream) {
  const float* q  = (const float*)d_in[0];
  const float* k  = (const float*)d_in[1];
  const float* v  = (const float*)d_in[2];
  const float* WQ = (const float*)d_in[3];
  const float* WK = (const float*)d_in[4];
  const float* WV = (const float*)d_in[5];
  float* out = (float*)d_out;

  const size_t WSZ = (size_t)DMODEL * DMODEL;                 // 1M elems
  const size_t HSZ = (size_t)BATCH * NHEAD * TSEQ * DHEAD;    // 8M elems
  // workspace layout (bf16): 3 transposed weights + Qh + Kh + VhT = ~54 MiB
  __bf16* WtQ = (__bf16*)d_ws;
  __bf16* WtK = WtQ + WSZ;
  __bf16* WtV = WtK + WSZ;
  __bf16* Qh  = WtV + WSZ;
  __bf16* Kh  = Qh + HSZ;
  __bf16* VhT = Kh + HSZ;

  mha_wt_transpose<<<dim3((unsigned)(WSZ / 256), 1, 3), 256, 0, stream>>>(
      WQ, WK, WV, WtQ, WtK, WtV);
  mha_qkv_proj<<<dim3(MROWS / 128, DMODEL / 128, 3), 256, 0, stream>>>(
      q, k, v, WtQ, WtK, WtV, Qh, Kh, VhT);
  mha_attention<<<dim3(TSEQ / 128, NHEAD, BATCH), 256, 0, stream>>>(
      Qh, Kh, VhT, q, out);
}
